// GCN_16088947490824
// MI455X (gfx1250) — matrix operand order repaired and checked by reading, not compiled, not run on hardware
//
#include <hip/hip_runtime.h>
#include <hip/hip_bf16.h>

// ---------------------------------------------------------------------------
// GCN forward: 3 x (GEMM+bias -> SPMM scatter-add) with ReLU/ReLU/Sigmoid.
// GEMMs in bf16 via v_wmma_f32_16x16x32_bf16. Each wave computes a 16x(16*NT)
// output strip: A-fragment loaded once per K-step, NT independent WMMA
// accumulator chains (hides WMMA->WMMA latency, cuts A re-read traffic NTx).
// SPMM via per-edge waves doing global_atomic_add_f32 into an L2-resident
// accumulator. MI455X: wave32, WMMA 16x16x32 bf16, 23.3 TB/s HBM, 192MB L2.
// ---------------------------------------------------------------------------

typedef __bf16 bf16_t;
typedef __attribute__((ext_vector_type(16))) __bf16 v16bf;
typedef __attribute__((ext_vector_type(8)))  __bf16 v8bf;
typedef __attribute__((ext_vector_type(8)))  float  v8f;

#define GN      50000
#define GE      1600000
#define F_IN    512
#define F_H1    256
#define F_H2    128
#define NCLS    40
#define NCLSP   48   // padded to 3 WMMA tiles

static __device__ __forceinline__ bf16_t f2bf(float f) {
    unsigned u = __builtin_bit_cast(unsigned, f);
    unsigned r = u + 0x7fffu + ((u >> 16) & 1u);   // RNE truncation
    unsigned short h = (unsigned short)(r >> 16);
    return __builtin_bit_cast(bf16_t, h);
}

// ---------------------------------------------------------------------------
// Pack row-major fp32 W[K x NoutSrc] into bf16 B-fragment-major layout:
//   Wp[((nt*Kt + kt)*32 + lane)*16 + i]
// lane<16:  col nt*16+lane,    K = kt*32 + (i<8 ? i : 16+i-8)
// lane>=16: col nt*16+lane-16, K = kt*32 + 8 + (i<8 ? i : 16+i-8)
// Columns >= NoutSrc are zero (padding for the 40->48 classifier).
// ---------------------------------------------------------------------------
__global__ void pack_w_kernel(const float* __restrict__ W, bf16_t* __restrict__ Wp,
                              int K, int NoutSrc, int NoutPad)
{
    int id = blockIdx.x * blockDim.x + threadIdx.x;
    int total = NoutPad * K;
    if (id >= total) return;
    int i    = id & 15;
    int lane = (id >> 4) & 31;
    int rest = id >> 9;            // nt*Kt + kt
    int Kt   = K >> 5;
    int kt   = rest % Kt;
    int nt   = rest / Kt;
    int n = nt * 16 + (lane & 15);
    int k = kt * 32 + ((i >> 3) << 4) + ((lane >> 4) << 3) + (i & 7);
    float w = (n < NoutSrc) ? W[(size_t)k * NoutSrc + n] : 0.0f;
    Wp[id] = f2bf(w);
}

__global__ void f32_to_bf16_kernel(const float* __restrict__ in,
                                   bf16_t* __restrict__ out, int n)
{
    int i = blockIdx.x * blockDim.x + threadIdx.x;
    if (i < n) out[i] = f2bf(in[i]);
}

__global__ void relu_to_bf16_kernel(const float* __restrict__ in,
                                    bf16_t* __restrict__ out, int n)
{
    int i = blockIdx.x * blockDim.x + threadIdx.x;
    if (i < n) out[i] = f2bf(fmaxf(in[i], 0.0f));
}

__global__ void zero_f32_kernel(float* __restrict__ p, int n)
{
    int i = blockIdx.x * blockDim.x + threadIdx.x;
    if (i < n) p[i] = 0.0f;
}

// ---------------------------------------------------------------------------
// GEMM: C[M x Nout] = A[M x K](bf16) @ Wp(packed bf16) + bias
// One wave per 16 x (16*NT) output strip; K consumed 32/step.
// A-fragment: two contiguous 16B loads per lane per K-step (shared by all NT
// column tiles). B-fragment: one contiguous 32B load per lane per tile.
// NT independent accumulators -> no WMMA->WMMA RAW hazard inside a K-step.
// Wave-uniform control flow only (EXEC must be all 1s for WMMA).
// ---------------------------------------------------------------------------
template <int NT>
__global__ __launch_bounds__(256)
void gemm_bf16_wmma_kernel(const bf16_t* __restrict__ A,
                           const bf16_t* __restrict__ Wp,
                           const float*  __restrict__ bias,
                           float* __restrict__ C,
                           int M, int K, int Nout, int nbias)
{
    const int lane = threadIdx.x & 31;
    const int wave = (blockIdx.x * blockDim.x + threadIdx.x) >> 5;
    const int Ngrp   = (Nout >> 4) / NT;          // groups of NT column tiles
    const int Mtiles = (M + 15) >> 4;
    if (wave >= Mtiles * Ngrp) return;            // uniform per wave
    const int tm = wave / Ngrp;
    const int tg = wave - tm * Ngrp;              // column-tile group
    const int Kt = K >> 5;

    int row  = tm * 16 + (lane & 15);
    int rowc = row < M ? row : M - 1;             // clamp, keep EXEC full
    const bf16_t* Ap = A + (size_t)rowc * K + ((lane >> 4) << 3);
    // Per-tile packed-B base pointers (tile tn = tg*NT + j)
    const bf16_t* Bp[NT];
    #pragma unroll
    for (int j = 0; j < NT; ++j)
        Bp[j] = Wp + ((size_t)(tg * NT + j) * Kt * 32 + lane) * 16;

    v8f acc[NT];
    #pragma unroll
    for (int j = 0; j < NT; ++j) acc[j] = v8f{};

    for (int kt = 0; kt < Kt; ++kt) {
        v8bf a0 = *(const v8bf*)(Ap + (kt << 5));
        v8bf a1 = *(const v8bf*)(Ap + (kt << 5) + 16);
        v16bf a = __builtin_shufflevector(a0, a1,
                    0,1,2,3,4,5,6,7,8,9,10,11,12,13,14,15);
        #pragma unroll
        for (int j = 0; j < NT; ++j) {
            v16bf b = *(const v16bf*)(Bp[j] + ((size_t)kt << 9));  // kt*32*16
            acc[j] = __builtin_amdgcn_wmma_f32_16x16x32_bf16(
                         false, a, false, b, (short)0, acc[j], false, false);
        }
    }

    const int r0 = tm * 16 + ((lane >> 4) << 3);  // C layout: vgpr v -> row r0+v
    #pragma unroll
    for (int j = 0; j < NT; ++j) {
        const int col = (tg * NT + j) * 16 + (lane & 15);
        const float bb = (col < nbias) ? bias[col] : 0.0f;
        #pragma unroll
        for (int v = 0; v < 8; ++v) {
            int r = r0 + v;
            if (r < M) C[(size_t)r * Nout + col] = acc[j][v] + bb;
        }
    }
}

// ---------------------------------------------------------------------------
// SPMM scatter-add: out[row[e], :] += val[e] * sup[col[e], :]
// One wave per edge; lanes stride the feature dimension. sup/out are
// L2-resident (<= 51 MB << 192 MB L2), atomics resolve at L2.
// ---------------------------------------------------------------------------
__global__ __launch_bounds__(256)
void spmm_edges_kernel(const int* __restrict__ row, const int* __restrict__ col,
                       const float* __restrict__ val,
                       const float* __restrict__ sup, float* __restrict__ out,
                       int F, int ld)
{
    int wave = (blockIdx.x * blockDim.x + threadIdx.x) >> 5;
    int lane = threadIdx.x & 31;
    if (wave >= GE) return;
    int s = col[wave];
    int d = row[wave];
    float v = val[wave];
    const float* sp = sup + (size_t)s * ld;
    float* op = out + (size_t)d * ld;
    for (int f = lane; f < F; f += 32)
        unsafeAtomicAdd(op + f, v * sp[f]);       // global_atomic_add_f32, no-return
}

__global__ void sigmoid_out_kernel(const float* __restrict__ g,
                                   float* __restrict__ out)
{
    int i = blockIdx.x * blockDim.x + threadIdx.x;
    if (i >= GN * NCLS) return;
    int n = i / NCLS;
    int f = i - n * NCLS;
    float x = g[(size_t)n * NCLSP + f];
    out[i] = 1.0f / (1.0f + __expf(-x));
}

// ---------------------------------------------------------------------------
// Workspace layout (bytes):
//   [0,       262144)     Wp0 packed bf16 512x256
//   [262144,  327680)     Wp1 packed bf16 256x128
//   [327680,  339968)     Wp2 packed bf16 128x48 (cols 40..47 zero)
//   [339968,  51539968)   bufA: bf16 activations, up to 50000x512
//   [51539968,102739968)  bufS: f32 GEMM output,  up to 50000x256
//   [102739968,153939968) bufG: f32 aggregate,    up to 50000x256
// ---------------------------------------------------------------------------
extern "C" void kernel_launch(void* const* d_in, const int* in_sizes, int n_in,
                              void* d_out, int out_size, void* d_ws, size_t ws_size,
                              hipStream_t stream)
{
    const float* x   = (const float*)d_in[0];
    const int*   row = (const int*)  d_in[1];
    const int*   col = (const int*)  d_in[2];
    const float* val = (const float*)d_in[3];
    const float* W0  = (const float*)d_in[4];
    const float* b0  = (const float*)d_in[5];
    const float* W1  = (const float*)d_in[6];
    const float* b1  = (const float*)d_in[7];
    const float* W2  = (const float*)d_in[8];
    const float* b2  = (const float*)d_in[9];
    float* out = (float*)d_out;

    char* ws = (char*)d_ws;
    bf16_t* Wp0  = (bf16_t*)(ws + 0);
    bf16_t* Wp1  = (bf16_t*)(ws + 262144);
    bf16_t* Wp2  = (bf16_t*)(ws + 327680);
    bf16_t* bufA = (bf16_t*)(ws + 339968);
    float*  bufS = (float*) (ws + 51539968);
    float*  bufG = (float*) (ws + 102739968);

    const int TPB = 256;
    auto cdiv = [](long long a, long long b) { return (int)((a + b - 1) / b); };

    // Weight packing + input conversion
    pack_w_kernel<<<cdiv(512 * 256, TPB), TPB, 0, stream>>>(W0, Wp0, 512, 256, 256);
    pack_w_kernel<<<cdiv(256 * 128, TPB), TPB, 0, stream>>>(W1, Wp1, 256, 128, 128);
    pack_w_kernel<<<cdiv(128 * NCLSP, TPB), TPB, 0, stream>>>(W2, Wp2, 128, NCLS, NCLSP);
    f32_to_bf16_kernel<<<cdiv((long long)GN * F_IN, TPB), TPB, 0, stream>>>(x, bufA, GN * F_IN);

    const int Mt = (GN + 15) / 16;
    const int spmm_blocks = cdiv((long long)GE * 32, TPB);

    // ---- Layer 0: 512 -> 256, ReLU ----  (16 col tiles = 4 groups of 4)
    {
        int waves = Mt * ((F_H1 / 16) / 4);
        gemm_bf16_wmma_kernel<4><<<cdiv((long long)waves * 32, TPB), TPB, 0, stream>>>(
            bufA, Wp0, b0, bufS, GN, F_IN, F_H1, F_H1);
        zero_f32_kernel<<<cdiv((long long)GN * F_H1, TPB), TPB, 0, stream>>>(bufG, GN * F_H1);
        spmm_edges_kernel<<<spmm_blocks, TPB, 0, stream>>>(row, col, val, bufS, bufG, F_H1, F_H1);
        relu_to_bf16_kernel<<<cdiv((long long)GN * F_H1, TPB), TPB, 0, stream>>>(bufG, bufA, GN * F_H1);
    }

    // ---- Layer 1: 256 -> 128, ReLU ----  (8 col tiles = 2 groups of 4)
    {
        int waves = Mt * ((F_H2 / 16) / 4);
        gemm_bf16_wmma_kernel<4><<<cdiv((long long)waves * 32, TPB), TPB, 0, stream>>>(
            bufA, Wp1, b1, bufS, GN, F_H1, F_H2, F_H2);
        zero_f32_kernel<<<cdiv((long long)GN * F_H2, TPB), TPB, 0, stream>>>(bufG, GN * F_H2);
        spmm_edges_kernel<<<spmm_blocks, TPB, 0, stream>>>(row, col, val, bufS, bufG, F_H2, F_H2);
        relu_to_bf16_kernel<<<cdiv((long long)GN * F_H2, TPB), TPB, 0, stream>>>(bufG, bufA, GN * F_H2);
    }

    // ---- Layer 2: 128 -> 40 (padded 48), Sigmoid ----  (3 col tiles, 1 group of 3)
    {
        int waves = Mt * 1;
        gemm_bf16_wmma_kernel<3><<<cdiv((long long)waves * 32, TPB), TPB, 0, stream>>>(
            bufA, Wp2, b2, bufS, GN, F_H2, NCLSP, NCLS);
        zero_f32_kernel<<<cdiv((long long)GN * NCLSP, TPB), TPB, 0, stream>>>(bufG, GN * NCLSP);
        spmm_edges_kernel<<<spmm_blocks, TPB, 0, stream>>>(row, col, val, bufS, bufG, NCLS, NCLSP);
        sigmoid_out_kernel<<<cdiv((long long)GN * NCLS, TPB), TPB, 0, stream>>>(bufG, out);
    }
}
